// Attention_11304353923779
// MI455X (gfx1250) — matrix operand order repaired and checked by reading, not compiled, not run on hardware
//
#include <hip/hip_runtime.h>
#include <hip/hip_bf16.h>

typedef __bf16 bf16_t;
typedef __attribute__((ext_vector_type(16))) __bf16 v16bf;
typedef __attribute__((ext_vector_type(8)))  __bf16 v8bf;
typedef __attribute__((ext_vector_type(8)))  float  v8f;

#define B_      4
#define N_      2048
#define DMODEL  1024
#define HEADS_  16
#define HDIM    64
#define TOKENS  (B_ * N_)         // 8192
#define SCALE_  0.03125f          // 1/sqrt(1024)

// ---------------------------------------------------------------------------
// WMMA fragment helpers (wave32, v_wmma_f32_16x16x32_bf16)
// A fragment 16x32 (M x K): lane%16 = M row; e=0..15 -> K = (e/8)*16 + (lane/16)*8 + e%8
// ---------------------------------------------------------------------------
__device__ __forceinline__ v16bf load_frag_a(const bf16_t* p, int ld) {
  const int lane = threadIdx.x & 31;
  const int m = lane & 15, half = lane >> 4;
  const bf16_t* base = p + (size_t)m * ld + half * 8;
  union { v16bf v; v8bf h[2]; } u;
  u.h[0] = *(const v8bf*)(base);
  u.h[1] = *(const v8bf*)(base + 16);
  return u.v;
}

// B fragment 32x16 (K x N), loaded from N-major (transposed) storage:
// lane%16 = N col; e=0..15 -> K = (e/8)*16 + (lane/16)*8 + e%8
__device__ __forceinline__ v16bf load_frag_b(const bf16_t* p, int ld) {
  const int lane = threadIdx.x & 31;
  const int n = lane & 15, half = lane >> 4;
  const bf16_t* base = p + (size_t)n * ld + half * 8;
  union { v16bf v; v8bf h[2]; } u;
  u.h[0] = *(const v8bf*)(base);
  u.h[1] = *(const v8bf*)(base + 16);
  return u.v;
}

__device__ __forceinline__ v8f wmma_bf16(v16bf a, v16bf b, v8f c) {
  return __builtin_amdgcn_wmma_f32_16x16x32_bf16(
      /*neg_a=*/false, a, /*neg_b=*/false, b,
      /*c_mod=*/(short)0, c, /*reuse_a=*/false, /*reuse_b=*/false);
}

// ---------------------------------------------------------------------------
// Kernel 1: fp32 -> bf16 elementwise conversion (for x)
// ---------------------------------------------------------------------------
__global__ void k_cvt_x(const float* __restrict__ x, bf16_t* __restrict__ xb, int n) {
  int i = blockIdx.x * blockDim.x + threadIdx.x;
  if (i < n) xb[i] = (bf16_t)x[i];
}

// ---------------------------------------------------------------------------
// Kernel 2: transpose + convert the four 1024x1024 weights: T[n][k] = W[k][n]
// ---------------------------------------------------------------------------
__global__ void k_transpose_w(const float* __restrict__ Wq, const float* __restrict__ Wk,
                              const float* __restrict__ Wv, const float* __restrict__ Wo,
                              bf16_t* __restrict__ Tq, bf16_t* __restrict__ Tk,
                              bf16_t* __restrict__ Tv, bf16_t* __restrict__ To) {
  const float* src; bf16_t* dst;
  switch (blockIdx.z) {
    case 0:  src = Wq; dst = Tq; break;
    case 1:  src = Wk; dst = Tk; break;
    case 2:  src = Wv; dst = Tv; break;
    default: src = Wo; dst = To; break;
  }
  int n = blockIdx.x * 16 + threadIdx.x;
  int k = blockIdx.y * 16 + threadIdx.y;
  dst[(size_t)n * DMODEL + k] = (bf16_t)src[(size_t)k * DMODEL + n];
}

// ---------------------------------------------------------------------------
// Kernel 3: QKV projection GEMM (8192x1024 @ 1024x1024), bf16 WMMA, f32 acc.
// gridDim.z selects Q/K/V. Q,K stored [b][h][n][d]; V stored [b][h][d][n].
// Block = 128 threads = 4 waves; each wave computes a 64x64 tile (4x4
// fragments -> 16 WMMA per 8 fragment loads per k-step); WG tile = 128x128.
// ---------------------------------------------------------------------------
__global__ void k_gemm_qkv(const bf16_t* __restrict__ xb,
                           const bf16_t* __restrict__ Tq, const bf16_t* __restrict__ Tk,
                           const bf16_t* __restrict__ Tv,
                           bf16_t* __restrict__ Q, bf16_t* __restrict__ K,
                           bf16_t* __restrict__ V) {
  const int lane = threadIdx.x & 31;
  const int wave = threadIdx.x >> 5;
  const int tm = blockIdx.x * 128 + (wave >> 1) * 64;
  const int tn = blockIdx.y * 128 + (wave & 1) * 64;
  const bf16_t* WT = (blockIdx.z == 0) ? Tq : (blockIdx.z == 1) ? Tk : Tv;

  v8f c[4][4];
#pragma unroll
  for (int i = 0; i < 4; ++i)
#pragma unroll
    for (int j = 0; j < 4; ++j) c[i][j] = (v8f){};

  for (int k = 0; k < DMODEL; k += 32) {
    v16bf a[4], b[4];
#pragma unroll
    for (int i = 0; i < 4; ++i)
      a[i] = load_frag_a(xb + (size_t)(tm + 16 * i) * DMODEL + k, DMODEL);
#pragma unroll
    for (int j = 0; j < 4; ++j)
      b[j] = load_frag_b(WT + (size_t)(tn + 16 * j) * DMODEL + k, DMODEL);
#pragma unroll
    for (int i = 0; i < 4; ++i)
#pragma unroll
      for (int j = 0; j < 4; ++j)
        c[i][j] = wmma_bf16(a[i], b[j], c[i][j]);
  }

  bf16_t* dst = (blockIdx.z == 0) ? Q : (blockIdx.z == 1) ? K : V;
  const bool vtrans = (blockIdx.z == 2);
#pragma unroll
  for (int i = 0; i < 4; ++i)
#pragma unroll
    for (int j = 0; j < 4; ++j)
#pragma unroll
      for (int r = 0; r < 8; ++r) {
        int row = tm + i * 16 + r + 8 * (lane >> 4);   // token index 0..8191
        int col = tn + j * 16 + (lane & 15);           // channel 0..1023
        int bidx = row >> 11, t = row & (N_ - 1);
        int h = col >> 6,  d = col & (HDIM - 1);
        bf16_t val = (bf16_t)c[i][j][r];
        if (vtrans) dst[(((size_t)bidx * HEADS_ + h) * HDIM + d) * N_ + t] = val;
        else        dst[(((size_t)bidx * HEADS_ + h) * N_ + t) * HDIM + d] = val;
      }
}

// ---------------------------------------------------------------------------
// Kernel 4: flash attention. One wave handles 16 queries x full head (d=64),
// streaming 32-key tiles with online softmax. P re-layout C-frag -> A-frag
// goes through a per-wave LDS buffer (LDS ops are in-order within a wave).
// ---------------------------------------------------------------------------
__global__ void k_flash_attn(const bf16_t* __restrict__ Q, const bf16_t* __restrict__ K,
                             const bf16_t* __restrict__ Vt, bf16_t* __restrict__ AO) {
  __shared__ __align__(16) bf16_t lds_p[4][16 * 32];
  const int lane = threadIdx.x & 31;
  const int wave = threadIdx.x >> 5;
  const int bh = blockIdx.x;                 // 0..63
  const int b = bh >> 4, h = bh & 15;
  const int q0 = blockIdx.y * 64 + wave * 16;  // query row within [0,2048)

  const bf16_t* Qh = Q  + (size_t)bh * N_ * HDIM;   // [n][d]
  const bf16_t* Kh = K  + (size_t)bh * N_ * HDIM;   // [n][d]
  const bf16_t* Vh = Vt + (size_t)bh * HDIM * N_;   // [d][n]
  bf16_t* pbuf = lds_p[wave];

  // Q fragments held in registers for the whole key loop
  v16bf qa0 = load_frag_a(Qh + (size_t)q0 * HDIM + 0,  HDIM);
  v16bf qa1 = load_frag_a(Qh + (size_t)q0 * HDIM + 32, HDIM);

  float mrow[8], lrow[8];
  for (int r = 0; r < 8; ++r) { mrow[r] = -1e30f; lrow[r] = 0.0f; }
  v8f o0 = {}, o1 = {}, o2 = {}, o3 = {};

  for (int kt = 0; kt < N_; kt += 32) {
    // S = Q * K^T for 32 keys: two 16x16 fragments
    v8f s0 = {}, s1 = {};
    {
      v16bf kb0 = load_frag_b(Kh + (size_t)kt * HDIM + 0,  HDIM);
      v16bf kb1 = load_frag_b(Kh + (size_t)kt * HDIM + 32, HDIM);
      s0 = wmma_bf16(qa0, kb0, s0);
      s0 = wmma_bf16(qa1, kb1, s0);
      v16bf kb2 = load_frag_b(Kh + (size_t)(kt + 16) * HDIM + 0,  HDIM);
      v16bf kb3 = load_frag_b(Kh + (size_t)(kt + 16) * HDIM + 32, HDIM);
      s1 = wmma_bf16(qa0, kb2, s1);
      s1 = wmma_bf16(qa1, kb3, s1);
    }

    // online softmax per row (row M = r + 8*(lane/16); N = lane%16)
    for (int r = 0; r < 8; ++r) {
      float v0 = s0[r] * SCALE_, v1 = s1[r] * SCALE_;
      float mx = fmaxf(v0, v1);
      mx = fmaxf(mx, __shfl_xor(mx, 1, 32));
      mx = fmaxf(mx, __shfl_xor(mx, 2, 32));
      mx = fmaxf(mx, __shfl_xor(mx, 4, 32));
      mx = fmaxf(mx, __shfl_xor(mx, 8, 32));
      float mnew  = fmaxf(mrow[r], mx);
      float alpha = __expf(mrow[r] - mnew);
      float p0 = __expf(v0 - mnew);
      float p1 = __expf(v1 - mnew);
      float rs = p0 + p1;
      rs += __shfl_xor(rs, 1, 32);
      rs += __shfl_xor(rs, 2, 32);
      rs += __shfl_xor(rs, 4, 32);
      rs += __shfl_xor(rs, 8, 32);
      lrow[r] = lrow[r] * alpha + rs;
      mrow[r] = mnew;
      o0[r] *= alpha; o1[r] *= alpha; o2[r] *= alpha; o3[r] *= alpha;
      // stage P row-major 16x32 in per-wave LDS buffer
      int prow = r + 8 * (lane >> 4);
      pbuf[prow * 32 + (lane & 15)]      = (bf16_t)p0;
      pbuf[prow * 32 + 16 + (lane & 15)] = (bf16_t)p1;
    }

    // O += P * V  (A = P 16x32 from LDS, B = Vt rows = d, contiguous keys)
    v16bf pa  = load_frag_a(pbuf, 32);
    v16bf vb0 = load_frag_b(Vh + (size_t)0  * N_ + kt, N_);
    v16bf vb1 = load_frag_b(Vh + (size_t)16 * N_ + kt, N_);
    v16bf vb2 = load_frag_b(Vh + (size_t)32 * N_ + kt, N_);
    v16bf vb3 = load_frag_b(Vh + (size_t)48 * N_ + kt, N_);
    o0 = wmma_bf16(pa, vb0, o0);
    o1 = wmma_bf16(pa, vb1, o1);
    o2 = wmma_bf16(pa, vb2, o2);
    o3 = wmma_bf16(pa, vb3, o3);
  }

  // normalize and store to [token][h*64 + d] (bf16), ready for the Wo GEMM
  for (int r = 0; r < 8; ++r) {
    int row_local = q0 + r + 8 * (lane >> 4);
    size_t token = (size_t)b * N_ + row_local;
    float inv = 1.0f / lrow[r];
    size_t base = token * DMODEL + (size_t)h * HDIM + (lane & 15);
    AO[base + 0]  = (bf16_t)(o0[r] * inv);
    AO[base + 16] = (bf16_t)(o1[r] * inv);
    AO[base + 32] = (bf16_t)(o2[r] * inv);
    AO[base + 48] = (bf16_t)(o3[r] * inv);
  }
}

// ---------------------------------------------------------------------------
// Kernel 5: output projection GEMM + bias, fp32 result (64x64 wave tile)
// ---------------------------------------------------------------------------
__global__ void k_gemm_out(const bf16_t* __restrict__ AO, const bf16_t* __restrict__ To,
                           const float* __restrict__ bo, float* __restrict__ out) {
  const int lane = threadIdx.x & 31;
  const int wave = threadIdx.x >> 5;
  const int tm = blockIdx.x * 128 + (wave >> 1) * 64;
  const int tn = blockIdx.y * 128 + (wave & 1) * 64;

  v8f c[4][4];
#pragma unroll
  for (int i = 0; i < 4; ++i)
#pragma unroll
    for (int j = 0; j < 4; ++j) c[i][j] = (v8f){};

  for (int k = 0; k < DMODEL; k += 32) {
    v16bf a[4], b[4];
#pragma unroll
    for (int i = 0; i < 4; ++i)
      a[i] = load_frag_a(AO + (size_t)(tm + 16 * i) * DMODEL + k, DMODEL);
#pragma unroll
    for (int j = 0; j < 4; ++j)
      b[j] = load_frag_b(To + (size_t)(tn + 16 * j) * DMODEL + k, DMODEL);
#pragma unroll
    for (int i = 0; i < 4; ++i)
#pragma unroll
      for (int j = 0; j < 4; ++j)
        c[i][j] = wmma_bf16(a[i], b[j], c[i][j]);
  }

#pragma unroll
  for (int i = 0; i < 4; ++i)
#pragma unroll
    for (int j = 0; j < 4; ++j) {
      int col = tn + j * 16 + (lane & 15);
      float bias = bo[col];
#pragma unroll
      for (int r = 0; r < 8; ++r) {
        int row = tm + i * 16 + r + 8 * (lane >> 4);
        out[(size_t)row * DMODEL + col] = c[i][j][r] + bias;
      }
    }
}

// ---------------------------------------------------------------------------
extern "C" void kernel_launch(void* const* d_in, const int* in_sizes, int n_in,
                              void* d_out, int out_size, void* d_ws, size_t ws_size,
                              hipStream_t stream) {
  (void)in_sizes; (void)n_in; (void)out_size; (void)ws_size;
  const float* x  = (const float*)d_in[0];
  const float* Wq = (const float*)d_in[1];
  const float* Wk = (const float*)d_in[2];
  const float* Wv = (const float*)d_in[3];
  const float* Wo = (const float*)d_in[4];
  const float* bo = (const float*)d_in[5];
  float* out = (float*)d_out;

  // workspace carve-up (bf16): xb 16MB | 4x weightT 8MB | Q,K,Vt 48MB | AO 16MB
  bf16_t* p  = (bf16_t*)d_ws;
  bf16_t* xb = p;  p += (size_t)TOKENS * DMODEL;
  bf16_t* Tq = p;  p += (size_t)DMODEL * DMODEL;
  bf16_t* Tk = p;  p += (size_t)DMODEL * DMODEL;
  bf16_t* Tv = p;  p += (size_t)DMODEL * DMODEL;
  bf16_t* To = p;  p += (size_t)DMODEL * DMODEL;
  bf16_t* Qm = p;  p += (size_t)TOKENS * DMODEL;
  bf16_t* Km = p;  p += (size_t)TOKENS * DMODEL;
  bf16_t* Vt = p;  p += (size_t)TOKENS * DMODEL;
  bf16_t* AO = p;  p += (size_t)TOKENS * DMODEL;

  k_cvt_x<<<(TOKENS * DMODEL) / 256, 256, 0, stream>>>(x, xb, TOKENS * DMODEL);
  k_transpose_w<<<dim3(DMODEL / 16, DMODEL / 16, 4), dim3(16, 16), 0, stream>>>(
      Wq, Wk, Wv, Wo, Tq, Tk, Tv, To);
  k_gemm_qkv<<<dim3(TOKENS / 128, DMODEL / 128, 3), 128, 0, stream>>>(
      xb, Tq, Tk, Tv, Qm, Km, Vt);
  k_flash_attn<<<dim3(B_ * HEADS_, N_ / 64), 128, 0, stream>>>(Qm, Km, Vt, AO);
  k_gemm_out<<<dim3(TOKENS / 128, DMODEL / 128), 128, 0, stream>>>(AO, To, bo, out);
}
